// LengthConditionedCrosswordSolver_14740327759961
// MI455X (gfx1250) — matrix operand order repaired and checked by reading, not compiled, not run on hardware
//
#include <hip/hip_runtime.h>
#include <hip/hip_bf16.h>

#define BATCH 8
#define SEQ   128
#define HID   512
#define VOC   32000
#define LMAX  20

typedef __attribute__((ext_vector_type(16))) __bf16 v16bf;
typedef __attribute__((ext_vector_type(8)))  __bf16 v8bf;
typedef __attribute__((ext_vector_type(4)))  __bf16 v4bf;
typedef __attribute__((ext_vector_type(8)))  float  v8f;

__device__ __forceinline__ __bf16 to_bf16(float f) { return (__bf16)f; }

// ---------------------------------------------------------------------------
// Phase 1: linfo[b][h] = dot(length_table[idx[b]], proj_W[h]) + proj_b[h]
// ---------------------------------------------------------------------------
__global__ __launch_bounds__(256)
void k_linfo(const int* __restrict__ alen, const float* __restrict__ ltab,
             const float* __restrict__ pW, const float* __restrict__ pb,
             float* __restrict__ linfo) {
    const int b   = blockIdx.x;
    const int idx = min(max(alen[b], 1), LMAX);
    const float* __restrict__ e = ltab + (size_t)idx * HID;
    for (int h = threadIdx.x; h < HID; h += blockDim.x) {
        const float* __restrict__ w = pW + (size_t)h * HID;
        float s = 0.f;
        for (int k = 0; k < HID; k += 4) {
            float4 ev = *(const float4*)(e + k);
            float4 wv = *(const float4*)(w + k);
            s += ev.x * wv.x + ev.y * wv.y + ev.z * wv.z + ev.w * wv.w;
        }
        linfo[b * HID + h] = s + pb[h];
    }
}

// ---------------------------------------------------------------------------
// Phase 2: enhanced_bf16[b][s][h] = bf16(hidden[b][s][h] + linfo[b][h])
// ---------------------------------------------------------------------------
__global__ __launch_bounds__(256)
void k_enh(const float* __restrict__ hidden, const float* __restrict__ linfo,
           unsigned short* __restrict__ enh) {
    const size_t i = ((size_t)blockIdx.x * blockDim.x + threadIdx.x) * 4;
    const int b = (int)(i >> 16);          // S*H == 65536
    const int h = (int)(i & (HID - 1));
    float4 hv = *(const float4*)(hidden + i);
    float4 lv = *(const float4*)(linfo + b * HID + h);
    v4bf o;
    o[0] = to_bf16(hv.x + lv.x);
    o[1] = to_bf16(hv.y + lv.y);
    o[2] = to_bf16(hv.z + lv.z);
    o[3] = to_bf16(hv.w + lv.w);
    *(v4bf*)(enh + i) = o;
}

// ---------------------------------------------------------------------------
// Phase 3: logits[b][s][v] = enhanced[b][s][:] . heads_W[idx[b]][v][:] + bias
// 4 waves / block; each wave computes a 16(M) x 64(N) tile, K-loop over H
// unrolled x2 (K=64 per trip) so WMMA->VALU hazard slots fill with the other
// half-iteration's loads/converts instead of v_nop.
// ---------------------------------------------------------------------------
__device__ __forceinline__ v16bf load_bfrag(const float* __restrict__ p) {
    const float4* pw = (const float4*)p;
    float4 w0 = pw[0], w1 = pw[1], w2 = pw[2], w3 = pw[3];
    v16bf bfv;
    bfv[0]  = to_bf16(w0.x); bfv[1]  = to_bf16(w0.y);
    bfv[2]  = to_bf16(w0.z); bfv[3]  = to_bf16(w0.w);
    bfv[4]  = to_bf16(w1.x); bfv[5]  = to_bf16(w1.y);
    bfv[6]  = to_bf16(w1.z); bfv[7]  = to_bf16(w1.w);
    bfv[8]  = to_bf16(w2.x); bfv[9]  = to_bf16(w2.y);
    bfv[10] = to_bf16(w2.z); bfv[11] = to_bf16(w2.w);
    bfv[12] = to_bf16(w3.x); bfv[13] = to_bf16(w3.y);
    bfv[14] = to_bf16(w3.z); bfv[15] = to_bf16(w3.w);
    return bfv;
}

__device__ __forceinline__ v16bf load_afrag(const unsigned short* __restrict__ p) {
    const v8bf* pa = (const v8bf*)p;
    v8bf alo = pa[0];           // K chunk at +0
    v8bf ahi = pa[1];           // K chunk at +8 elements
    v16bf af;
#pragma unroll
    for (int i = 0; i < 8; ++i) { af[i] = alo[i]; af[i + 8] = ahi[i]; }
    return af;
}

__global__ __launch_bounds__(128)
void k_gemm(const unsigned short* __restrict__ enh,
            const int* __restrict__ alen,
            const float* __restrict__ headsW,   // [L+1, V, H]
            const float* __restrict__ headsB,   // [L+1, V]
            float* __restrict__ out) {          // [B, S, V]
    const int lane = threadIdx.x & 31;
    const int wave = threadIdx.x >> 5;
    const int b    = blockIdx.z;
    const int m0   = blockIdx.y * 16;
    const int n0   = (blockIdx.x * 4 + wave) * 64;

    const int idx = min(max(alen[b], 1), LMAX);
    const float* __restrict__ Wsel = headsW + (size_t)idx * VOC * HID;
    const float* __restrict__ Bsel = headsB + (size_t)idx * VOC;

    const int nl   = lane & 15;   // row-in-tile (A) / N column (B,C)
    const int half = lane >> 4;   // lane-half selector

    // A fragment source: row (m0+nl), K chunks at {half*16, half*16+8} per step
    const unsigned short* __restrict__ Arow =
        enh + ((size_t)b * SEQ + m0 + nl) * HID + half * 16;

    // B fragment sources: 4 N-tiles, each lane owns a contiguous 16-float chunk
    const float* __restrict__ wrows[4];
#pragma unroll
    for (int t = 0; t < 4; ++t)
        wrows[t] = Wsel + (size_t)(n0 + t * 16 + nl) * HID + half * 16;

    v8f acc[4] = {};

    for (int k0 = 0; k0 < HID; k0 += 64) {     // 8 trips, 2 K-steps each
        v16bf af0 = load_afrag(Arow + k0);
        v16bf af1 = load_afrag(Arow + k0 + 32);
#pragma unroll
        for (int t = 0; t < 4; ++t) {
            v16bf b0 = load_bfrag(wrows[t] + k0);
            v16bf b1 = load_bfrag(wrows[t] + k0 + 32);
            acc[t] = __builtin_amdgcn_wmma_f32_16x16x32_bf16(
                false, af0, false, b0, (short)0, acc[t], false, false);
            acc[t] = __builtin_amdgcn_wmma_f32_16x16x32_bf16(
                false, af1, false, b1, (short)0, acc[t], false, false);
        }
    }

    // ---- Epilogue: bias add + store (C layout: VGPR r -> M=r / r+8) ----
#pragma unroll
    for (int t = 0; t < 4; ++t) {
        const int gcol = n0 + t * 16 + nl;
        const float bias = Bsel[gcol];
#pragma unroll
        for (int r = 0; r < 8; ++r) {
            out[((size_t)b * SEQ + m0 + half * 8 + r) * VOC + gcol] =
                acc[t][r] + bias;
        }
    }
}

// ---------------------------------------------------------------------------
extern "C" void kernel_launch(void* const* d_in, const int* in_sizes, int n_in,
                              void* d_out, int out_size, void* d_ws, size_t ws_size,
                              hipStream_t stream) {
    const float* hidden = (const float*)d_in[0];   // [B,S,H]
    const int*   alen   = (const int*)  d_in[1];   // [B]
    const float* ltab   = (const float*)d_in[2];   // [L+1,H]
    const float* pW     = (const float*)d_in[3];   // [H,H]
    const float* pb     = (const float*)d_in[4];   // [H]
    const float* headsW = (const float*)d_in[5];   // [L+1,V,H]
    const float* headsB = (const float*)d_in[6];   // [L+1,V]
    float* out = (float*)d_out;

    float*          linfo = (float*)d_ws;                           // 16 KB
    unsigned short* enh   = (unsigned short*)((char*)d_ws + 16384); // 1 MB bf16

    k_linfo<<<BATCH, 256, 0, stream>>>(alen, ltab, pW, pb, linfo);

    const int enh_blocks = (BATCH * SEQ * HID) / (4 * 256);         // 512
    k_enh<<<enh_blocks, 256, 0, stream>>>(hidden, linfo, enh);

    dim3 grid(VOC / (4 * 64), SEQ / 16, BATCH);                     // 125 x 8 x 8
    k_gemm<<<grid, 128, 0, stream>>>(enh, alen, headsW, headsB, out);
}
